// DendriticFullyConnected_68178310856894
// MI455X (gfx1250) — compile-verified, matrix operand order checked by reading, not run on hardware
//
#include <hip/hip_runtime.h>
#include <hip/hip_bf16.h>
#include <math.h>

// ---------------------------------------------------------------------------
// Problem constants
// ---------------------------------------------------------------------------
#define M_DIM      8192          // BATCH
#define N_DIM      4096          // OUT_FEATURES
#define K_IN       4096          // IN_FEATURES
#define IC         409           // int(4096*0.1)
#define K1_PAD     416           // IC padded to 13*32
#define K2         3687          // IN - IC
#define K2_PAD     3712          // padded to 116*32
#define K_TOT      4128          // K1_PAD + K2_PAD  (129 tiles of 32)
#define NK_TILES   129
#define KSPLIT_T   13            // first 13 K-tiles -> nmda accumulator

#define BM         128
#define BN         64
#define ROWP4      5             // padded LDS row: 40 bf16 = 5 uint4 (80 B)

// per-buffer uint4 offsets in LDS
#define AHI_OFF    0             // 128*5 = 640
#define ALO_OFF    640
#define BHI_OFF    1280          // 64*5  = 320
#define BLO_OFF    1600
#define BUF_U4     1920          // per double-buffer half

typedef __attribute__((ext_vector_type(16))) __bf16 v16bf;
typedef __attribute__((ext_vector_type(8)))  float  v8f;

union FragU {
    uint4 q[2];
    v16bf v;
};

// ---------------------------------------------------------------------------
// CDNA5 async global->LDS via inline asm (bypasses builtin AS typing).
// GV addressing: 64-bit global address in a VGPR pair, VDST = LDS offset.
// Tracked by ASYNCcnt; completion enforced with s_wait_asynccnt.
// ---------------------------------------------------------------------------
__device__ __forceinline__ void async_load_b128(const void* gsrc, void* ldst) {
    unsigned lds_off = (unsigned)(size_t)ldst;               // addr[31:0] = LDS offset
    unsigned long long gaddr = (unsigned long long)(size_t)gsrc;
    asm volatile("global_load_async_to_lds_b128 %0, %1, off"
                 :: "v"(lds_off), "v"(gaddr)
                 : "memory");
}
__device__ __forceinline__ void wait_async_then_barrier() {
    asm volatile("s_wait_asynccnt 0x0" ::: "memory");
    __syncthreads();
}

// round-to-nearest-even fp32 -> bf16 (as ushort), and back
__device__ __forceinline__ unsigned short f2bf(float f) {
    unsigned u = __float_as_uint(f);
    u += 0x7FFFu + ((u >> 16) & 1u);
    return (unsigned short)(u >> 16);
}
__device__ __forceinline__ float bf2f(unsigned short h) {
    return __uint_as_float(((unsigned)h) << 16);
}

// ---------------------------------------------------------------------------
// Pack x into combined-K, coeff-scaled, zero-padded hi/lo bf16
//   col kp <  416 : src = kp<409 ? x[r,kp]*coeff(kp) : 0
//   col kp >= 416 : src = kp<4103 ? x[r, kp-7] : 0
// ---------------------------------------------------------------------------
__global__ void pack_x_kernel(const float* __restrict__ x,
                              unsigned short* __restrict__ xhi,
                              unsigned short* __restrict__ xlo) {
    int kp = blockIdx.x * blockDim.x + threadIdx.x;
    int r  = blockIdx.y;
    if (kp >= K_TOT) return;
    float v = 0.0f;
    if (kp < K1_PAD) {
        if (kp < IC) {
            float c = (kp == 0 || kp == IC - 1) ? 1.0f : 2.0f;
            v = x[(size_t)r * K_IN + kp] * c;
        }
    } else {
        if (kp < K1_PAD + K2) v = x[(size_t)r * K_IN + (kp - 7)];
    }
    unsigned short hi = f2bf(v);
    float lo = v - bf2f(hi);
    size_t o = (size_t)r * K_TOT + kp;
    xhi[o] = hi;
    xlo[o] = f2bf(lo);
}

// ---------------------------------------------------------------------------
// Pack [W_nmda | W_non] into combined-K zero-padded hi/lo bf16 (N-major)
// ---------------------------------------------------------------------------
__global__ void pack_w_kernel(const float* __restrict__ wn,
                              const float* __restrict__ wo,
                              unsigned short* __restrict__ whi,
                              unsigned short* __restrict__ wlo) {
    int kp = blockIdx.x * blockDim.x + threadIdx.x;
    int n  = blockIdx.y;
    if (kp >= K_TOT) return;
    float v = 0.0f;
    if (kp < K1_PAD) {
        if (kp < IC) v = wn[(size_t)n * IC + kp];
    } else {
        if (kp < K1_PAD + K2) v = wo[(size_t)n * K2 + (kp - K1_PAD)];
    }
    unsigned short hi = f2bf(v);
    float lo = v - bf2f(hi);
    size_t o = (size_t)n * K_TOT + kp;
    whi[o] = hi;
    wlo[o] = f2bf(lo);
}

// ---------------------------------------------------------------------------
// Fused dual-accumulator GEMM + epilogue.
// Block tile 128x64, 8 wave32 waves in a 4(m) x 2(n) grid, each wave 32x32
// (2x2 WMMA 16x16 tiles), bf16x3 split-precision accumulation in fp32.
// Two separate K phases (no per-iteration accumulator select).
// ---------------------------------------------------------------------------
__global__ __launch_bounds__(256)
void dendritic_gemm_kernel(const unsigned short* __restrict__ xhi,
                           const unsigned short* __restrict__ xlo,
                           const unsigned short* __restrict__ whi,
                           const unsigned short* __restrict__ wlo,
                           const float* __restrict__ bnon,
                           float* __restrict__ out) {
    __shared__ uint4 lds4[2 * BUF_U4];   // 61440 bytes

    const int t      = threadIdx.x;
    const int lane   = t & 31;
    const int wid    = t >> 5;
    const int wave_m = wid >> 1;         // 0..3
    const int wave_n = wid & 1;          // 0..1
    const int fr     = lane & 15;        // tile row (A) / tile col (B)
    const int kc     = lane >> 4;        // 16-bit layout chunk select

    const int blockM = blockIdx.y * BM;
    const int blockN = blockIdx.x * BN;

    const uint4* xhi4 = (const uint4*)xhi;
    const uint4* xlo4 = (const uint4*)xlo;
    const uint4* whi4 = (const uint4*)whi;
    const uint4* wlo4 = (const uint4*)wlo;
    const int    ld4  = K_TOT / 8;       // 516 uint4 per row

    v8f accA[2][2];  // nmda accumulator
    v8f accB[2][2];  // non-nmda accumulator
#pragma unroll
    for (int mi = 0; mi < 2; ++mi)
#pragma unroll
        for (int ni = 0; ni < 2; ++ni) {
            accA[mi][ni] = (v8f){};
            accB[mi][ni] = (v8f){};
        }

    // global -> LDS stage for one 32-wide K tile (async copies)
    auto load_tiles = [&](int buf, int kt) {
        uint4* L = &lds4[buf * BUF_U4];
        // A: 128 rows x 4 chunks = 512 chunks, 2 per thread
#pragma unroll
        for (int it = 0; it < 2; ++it) {
            int c   = t + it * 256;
            int row = c >> 2, seg = c & 3;
            size_t g = (size_t)(blockM + row) * ld4 + (size_t)kt * 4 + seg;
            int lo = row * ROWP4 + seg;
            async_load_b128(xhi4 + g, &L[AHI_OFF + lo]);
            async_load_b128(xlo4 + g, &L[ALO_OFF + lo]);
        }
        // B: 64 rows x 4 chunks = 256 chunks, 1 per thread
        {
            int row = t >> 2, seg = t & 3;
            size_t g = (size_t)(blockN + row) * ld4 + (size_t)kt * 4 + seg;
            int lo = row * ROWP4 + seg;
            async_load_b128(whi4 + g, &L[BHI_OFF + lo]);
            async_load_b128(wlo4 + g, &L[BLO_OFF + lo]);
        }
    };

    // one K-step: prefetch kt+1, compute kt into the given accumulator
    auto kstep = [&](int kt, v8f (&acc)[2][2]) {
        const int cur = kt & 1;
        if (kt + 1 < NK_TILES) load_tiles(cur ^ 1, kt + 1);

        const uint4* L = &lds4[cur * BUF_U4];

        // fragments: lane fr = row, kc selects k-chunks {0..7,16..23} vs
        // {8..15,24..31}; each chunk is one uint4 (16 bytes)
        FragU ah[2], al[2], bh[2], bl[2];
#pragma unroll
        for (int mi = 0; mi < 2; ++mi) {
            int r    = wave_m * 32 + mi * 16 + fr;
            int base = r * ROWP4 + kc;
            ah[mi].q[0] = L[AHI_OFF + base];
            ah[mi].q[1] = L[AHI_OFF + base + 2];
            al[mi].q[0] = L[ALO_OFF + base];
            al[mi].q[1] = L[ALO_OFF + base + 2];
        }
#pragma unroll
        for (int ni = 0; ni < 2; ++ni) {
            int r    = wave_n * 32 + ni * 16 + fr;
            int base = r * ROWP4 + kc;
            bh[ni].q[0] = L[BHI_OFF + base];
            bh[ni].q[1] = L[BHI_OFF + base + 2];
            bl[ni].q[0] = L[BLO_OFF + base];
            bl[ni].q[1] = L[BLO_OFF + base + 2];
        }

#pragma unroll
        for (int mi = 0; mi < 2; ++mi)
#pragma unroll
            for (int ni = 0; ni < 2; ++ni) {
                v8f c = acc[mi][ni];
                c = __builtin_amdgcn_wmma_f32_16x16x32_bf16(
                        false, ah[mi].v, false, bh[ni].v, (short)0, c, false, false);
                c = __builtin_amdgcn_wmma_f32_16x16x32_bf16(
                        false, ah[mi].v, false, bl[ni].v, (short)0, c, false, false);
                c = __builtin_amdgcn_wmma_f32_16x16x32_bf16(
                        false, al[mi].v, false, bh[ni].v, (short)0, c, false, false);
                acc[mi][ni] = c;
            }

        wait_async_then_barrier();
    };

    load_tiles(0, 0);
    wait_async_then_barrier();

    // Phase 1: NMDA block (K-tiles 0..12)
    for (int kt = 0; kt < KSPLIT_T; ++kt) kstep(kt, accA);
    // Phase 2: non-NMDA block (K-tiles 13..128)
    for (int kt = KSPLIT_T; kt < NK_TILES; ++kt) kstep(kt, accB);

    // Epilogue: state = sigmoid(accB + b) - 1 ; pre = accA + state
    //           out = pre^2 / (0.25 + pre^2)
#pragma unroll
    for (int mi = 0; mi < 2; ++mi)
#pragma unroll
        for (int ni = 0; ni < 2; ++ni) {
            int col = blockN + wave_n * 32 + ni * 16 + fr;
            float bn = bnon[col];
#pragma unroll
            for (int e = 0; e < 8; ++e) {
                int row = blockM + wave_m * 32 + mi * 16 + e + kc * 8;
                float a = accA[mi][ni][e];
                float s = accB[mi][ni][e] + bn;
                float state = 1.0f / (1.0f + __expf(-s)) - 1.0f;
                float pre = a + state;
                float xn = pre * pre;
                out[(size_t)row * N_DIM + col] = xn / (0.25f + xn);
            }
        }
}

// ---------------------------------------------------------------------------
// Launch
// ---------------------------------------------------------------------------
extern "C" void kernel_launch(void* const* d_in, const int* in_sizes, int n_in,
                              void* d_out, int out_size, void* d_ws, size_t ws_size,
                              hipStream_t stream) {
    const float* x      = (const float*)d_in[0];
    const float* w_nmda = (const float*)d_in[1];
    const float* w_non  = (const float*)d_in[2];
    const float* b_non  = (const float*)d_in[3];
    float* out = (float*)d_out;

    // workspace layout (ushorts)
    unsigned short* ws = (unsigned short*)d_ws;
    const size_t X_ELEMS = (size_t)M_DIM * K_TOT;   // 33,816,576
    const size_t W_ELEMS = (size_t)N_DIM * K_TOT;   // 16,908,288
    unsigned short* xhi = ws;
    unsigned short* xlo = xhi + X_ELEMS;
    unsigned short* whi = xlo + X_ELEMS;
    unsigned short* wlo = whi + W_ELEMS;

    dim3 pblk(256);
    dim3 pxgrid((K_TOT + 255) / 256, M_DIM);
    dim3 pwgrid((K_TOT + 255) / 256, N_DIM);
    pack_x_kernel<<<pxgrid, pblk, 0, stream>>>(x, xhi, xlo);
    pack_w_kernel<<<pwgrid, pblk, 0, stream>>>(w_nmda, w_non, whi, wlo);

    dim3 gblk(256);
    dim3 ggrid(N_DIM / BN, M_DIM / BM);   // 64 x 64
    dendritic_gemm_kernel<<<ggrid, gblk, 0, stream>>>(xhi, xlo, whi, wlo, b_non, out);
}
